// Attention_1443109011682
// MI455X (gfx1250) — matrix operand order repaired and checked by reading, not compiled, not run on hardware
//
#include <hip/hip_runtime.h>

typedef __attribute__((ext_vector_type(16))) _Float16 v16h;
typedef __attribute__((ext_vector_type(8)))  _Float16 v8h;
typedef __attribute__((ext_vector_type(2)))  _Float16 v2h;
typedef __attribute__((ext_vector_type(8)))  float    v8f;
typedef __attribute__((ext_vector_type(4)))  float    v4f;

#define DEV __device__ __forceinline__

constexpr int Bc = 2, Sc = 2048, DIMc = 2048, Hc = 16, KVHc = 8, Dc = 128;
constexpr float SCALEc = 0.08838834764831845f; // 1/sqrt(128)

// ---------------- workspace layout (bytes) ----------------
constexpr size_t QF32_OFF = 0;
constexpr size_t QF32_SZ  = (size_t)Bc * Hc   * Sc * Dc * 4;
constexpr size_t KF32_OFF = QF32_OFF + QF32_SZ;
constexpr size_t KF32_SZ  = (size_t)Bc * KVHc * Sc * Dc * 4;
constexpr size_t QF16_OFF = KF32_OFF + KF32_SZ;
constexpr size_t QF16_SZ  = (size_t)Bc * Hc   * Sc * Dc * 2;
constexpr size_t KF16_OFF = QF16_OFF + QF16_SZ;
constexpr size_t KF16_SZ  = (size_t)Bc * KVHc * Sc * Dc * 2;
constexpr size_t VF16_OFF = KF16_OFF + KF16_SZ;
constexpr size_t VF16_SZ  = KF16_SZ;
constexpr size_t AO_OFF   = VF16_OFF + VF16_SZ;   // attn out f16 [B,S,H*D]

// ---------------- vectorized WMMA fragment loaders ----------------
DEV v16h combine16(v8h lo, v8h hi) {
  return __builtin_shufflevector(lo, hi, 0, 1, 2, 3, 4, 5, 6, 7,
                                 8, 9, 10, 11, 12, 13, 14, 15);
}

// 16 contiguous halves at a lane-specific pointer (B-transposed layout)
DEV v16h frag16(const _Float16* p) {
  v8h lo = *(const v8h*)p;
  v8h hi = *(const v8h*)(p + 8);
  return combine16(lo, hi);
}

// A 16x32 f16 fragment: lane row m = lane&15; 16 contiguous halves at k = half*16
DEV v16h loadA16(const _Float16* base, int ld, int lane) {
  int m = lane & 15, half = lane >> 4;
  const _Float16* p = base + (size_t)m * ld + half * 16;
  v8h lo = *(const v8h*)p;
  v8h hi = *(const v8h*)(p + 8);
  return combine16(lo, hi);
}

// A fragment from fp32 source (x matrix): 4x b128 loads + packed converts
DEV v16h loadA_f32v(const float* base, int ld, int lane) {
  int m = lane & 15, half = lane >> 4;
  const float* p = base + (size_t)m * ld + half * 16;
  v4f a0 = *(const v4f*)p;
  v4f a1 = *(const v4f*)(p + 4);
  v4f b0 = *(const v4f*)(p + 8);
  v4f b1 = *(const v4f*)(p + 12);
  v16h a;
#pragma unroll
  for (int i = 0; i < 4; ++i) {
    a[i]      = (_Float16)a0[i];
    a[4 + i]  = (_Float16)a1[i];
    a[8 + i]  = (_Float16)b0[i];
    a[12 + i] = (_Float16)b1[i];
  }
  return a;
}

DEV v8f wmma_f16(v16h a, v16h b, v8f c) {
  return __builtin_amdgcn_wmma_f32_16x16x32_f16(false, a, false, b, (short)0, c,
                                                false, false);
}

// ---- CDNA5 async global->LDS copy (16B per lane), tracked by ASYNCcnt ----
DEV void async_copy_b128(unsigned lds_off, const void* gptr) {
  asm volatile("global_load_async_to_lds_b128 %0, %1, off"
               :: "v"(lds_off), "v"(gptr) : "memory");
}

DEV void wait_async0() {
#if defined(__has_builtin)
#if __has_builtin(__builtin_amdgcn_s_wait_asynccnt)
  __builtin_amdgcn_s_wait_asynccnt(0);
  return;
#endif
#endif
  asm volatile("s_wait_asynccnt 0x0" ::: "memory");
}

// ==================== Kernel 1: fused QKV projection ====================
// Block: 128 rows x 128 cols; per-wave 16x128 (8 WMMA per 32-K step).
__global__ __launch_bounds__(256) void qkv_gemm_k(
    const float* __restrict__ x, const float* __restrict__ wq,
    const float* __restrict__ wk, const float* __restrict__ wv,
    float* __restrict__ qf, float* __restrict__ kf, _Float16* __restrict__ vh) {
  __shared__ __align__(16) _Float16 bs[2][128][40];   // W^T: [n][k]
  int lane = threadIdx.x, w = threadIdx.y;
  int tid  = w * 32 + lane;
  int mt   = blockIdx.x >> 5;           // 32 row-tiles of 128
  int nb   = blockIdx.x & 31;           // 32 col-tiles of 128
  int rowW = mt * 128 + w * 16;
  int colb = nb * 128;

  const float* wp; int ldw, lc0;
  if (colb < 2048)      { wp = wq; ldw = 2048; lc0 = colb; }
  else if (colb < 3072) { wp = wk; ldw = 1024; lc0 = colb - 2048; }
  else                  { wp = wv; ldw = 1024; lc0 = colb - 3072; }

  int skk = (tid >> 5) * 4;             // 4 k-rows per thread
  int sn4 = (tid & 31) * 4;             // 4 n-cols per thread
  auto stage = [&](int k0, int p) {
#pragma unroll
    for (int rp = 0; rp < 2; ++rp) {
      int kr = skk + rp * 2;
      v4f r0 = *(const v4f*)(wp + (size_t)(k0 + kr) * ldw + lc0 + sn4);
      v4f r1 = *(const v4f*)(wp + (size_t)(k0 + kr + 1) * ldw + lc0 + sn4);
#pragma unroll
      for (int i = 0; i < 4; ++i) {
        v2h pk = { (_Float16)r0[i], (_Float16)r1[i] };
        *(v2h*)&bs[p][sn4 + i][kr] = pk;   // packed b32 store, transposed
      }
    }
  };

  v8f acc[8] = {};
  stage(0, 0);
  __syncthreads();
  int p = 0;
  int n = lane & 15, half = lane >> 4;
  for (int k0 = 0; k0 < DIMc; k0 += 32) {
    if (k0 + 32 < DIMc) stage(k0 + 32, p ^ 1);
    v16h a = loadA_f32v(x + (size_t)rowW * DIMc + k0, DIMc, lane);
    v16h bf[8];
#pragma unroll
    for (int t = 0; t < 8; ++t) bf[t] = frag16(&bs[p][t * 16 + n][half * 16]);
#pragma unroll
    for (int t = 0; t < 8; ++t) acc[t] = wmma_f16(a, bf[t], acc[t]);
    __syncthreads();
    p ^= 1;
  }

  if (colb < 2048) {
#pragma unroll
    for (int t = 0; t < 8; ++t) {
      int col = colb + t * 16 + n, h = col >> 7, d = col & 127;
#pragma unroll
      for (int r = 0; r < 8; ++r) {
        int row = rowW + r + 8 * half;
        int b = row >> 11, s = row & (Sc - 1);
        qf[(((size_t)b * Hc + h) * Sc + s) * Dc + d] = acc[t][r];
      }
    }
  } else if (colb < 3072) {
#pragma unroll
    for (int t = 0; t < 8; ++t) {
      int col = colb - 2048 + t * 16 + n, h = col >> 7, d = col & 127;
#pragma unroll
      for (int r = 0; r < 8; ++r) {
        int row = rowW + r + 8 * half;
        int b = row >> 11, s = row & (Sc - 1);
        kf[(((size_t)b * KVHc + h) * Sc + s) * Dc + d] = acc[t][r];
      }
    }
  } else {
#pragma unroll
    for (int t = 0; t < 8; ++t) {
      int col = colb - 3072 + t * 16 + n, h = col >> 7, d = col & 127;
#pragma unroll
      for (int r = 0; r < 8; ++r) {
        int row = rowW + r + 8 * half;
        int b = row >> 11, s = row & (Sc - 1);
        vh[(((size_t)b * KVHc + h) * Sc + s) * Dc + d] = (_Float16)acc[t][r];
      }
    }
  }
}

// ==================== Kernel 2: RMSNorm + RoPE (q and k) ====================
__global__ __launch_bounds__(256) void norm_rope_k(
    const float* __restrict__ qf, const float* __restrict__ kf,
    const float* __restrict__ qg, const float* __restrict__ kg,
    const float* __restrict__ cosc, const float* __restrict__ sinc,
    _Float16* __restrict__ qh, _Float16* __restrict__ kh) {
  int lane = threadIdx.x;
  int wid  = blockIdx.x * 8 + threadIdx.y;
  const int HH = Hc + KVHc;
  int hh = wid % HH, tok = wid / HH;
  int b = tok >> 11, s = tok & (Sc - 1);

  const float* in; const float* g; _Float16* outp;
  if (hh < Hc) {
    size_t off = (((size_t)b * Hc + hh) * Sc + s) * Dc;
    in = qf + off; g = qg; outp = qh + off;
  } else {
    size_t off = (((size_t)b * KVHc + (hh - Hc)) * Sc + s) * Dc;
    in = kf + off; g = kg; outp = kh + off;
  }

  float t0 = in[lane], t1 = in[lane + 32], t2 = in[lane + 64], t3 = in[lane + 96];
  float ss = t0 * t0 + t1 * t1 + t2 * t2 + t3 * t3;
#pragma unroll
  for (int xm = 1; xm < 32; xm <<= 1) ss += __shfl_xor(ss, xm, 32);
  float inv = rsqrtf(ss * (1.0f / 128.0f) + 1e-6f);

  float n0 = t0 * inv * g[lane];
  float n1 = t1 * inv * g[lane + 32];
  float n2 = t2 * inv * g[lane + 64];
  float n3 = t3 * inv * g[lane + 96];

  float c0 = cosc[(size_t)s * 64 + lane],      s0 = sinc[(size_t)s * 64 + lane];
  float c1 = cosc[(size_t)s * 64 + lane + 32], s1 = sinc[(size_t)s * 64 + lane + 32];
  float o0 = n0 * c0 - n2 * s0;
  float o2 = n2 * c0 + n0 * s0;
  float o1 = n1 * c1 - n3 * s1;
  float o3 = n3 * c1 + n1 * s1;

  outp[lane]      = (_Float16)o0;
  outp[lane + 32] = (_Float16)o1;
  outp[lane + 64] = (_Float16)o2;
  outp[lane + 96] = (_Float16)o3;
}

// ==================== Kernel 3: causal flash attention (GQA) ====================
// Block = 128 query rows of one (b,h); K staged via async global->LDS copies,
// V staged transposed through VGPRs; double-buffered; shared by 8 waves.
__global__ __launch_bounds__(256) void attn_k(
    const _Float16* __restrict__ qh, const _Float16* __restrict__ kh,
    const _Float16* __restrict__ vh, _Float16* __restrict__ ao) {
  __shared__ __align__(16) _Float16 ks[2][32][136];   // K natural [kv][d]
  __shared__ __align__(16) _Float16 vs[2][128][40];   // V transposed [d][kv]
  __shared__ __align__(16) _Float16 pbuf[8][16][40];  // per-wave P re-fragment

  int lane = threadIdx.x, w = threadIdx.y;
  int tid  = w * 32 + lane;
  int qt = blockIdx.x, h = blockIdx.y, b = blockIdx.z;
  int srowB = qt * 128;
  int srow0 = srowB + w * 16;
  int half = lane >> 4, n = lane & 15;

  const _Float16* Qb = qh + ((size_t)b * Hc + h) * Sc * Dc;
  int kvh = h >> 1;
  const _Float16* Kb = kh + ((size_t)b * KVHc + kvh) * Sc * Dc;
  const _Float16* Vb = vh + ((size_t)b * KVHc + kvh) * Sc * Dc;

  int skk = (tid >> 4) * 2;     // kv row pair: 0,2,..,30
  int sc8 = (tid & 15) * 8;     // d chunk: 0..120
  auto stage = [&](int kb, int p) {
    // K: async DMA straight into LDS (ASYNCcnt); rows are 272B apart
    unsigned l0 = (unsigned)(size_t)&ks[p][skk][sc8];
    async_copy_b128(l0,       Kb + (size_t)(kb + skk) * Dc + sc8);
    async_copy_b128(l0 + 272, Kb + (size_t)(kb + skk + 1) * Dc + sc8);
    // V: packed-pair transposed store via VGPRs
    v8h va = *(const v8h*)(Vb + (size_t)(kb + skk) * Dc + sc8);
    v8h vb = *(const v8h*)(Vb + (size_t)(kb + skk + 1) * Dc + sc8);
#pragma unroll
    for (int i = 0; i < 8; ++i) {
      v2h pk = { va[i], vb[i] };
      *(v2h*)&vs[p][sc8 + i][skk] = pk;
    }
  };

  v16h qa[4];
#pragma unroll
  for (int c = 0; c < 4; ++c)
    qa[c] = loadA16(Qb + (size_t)srow0 * Dc + c * 32, Dc, lane);

  v8f acc[8] = {};
  float m8[8], l8[8];
#pragma unroll
  for (int r = 0; r < 8; ++r) { m8[r] = -3.0e38f; l8[r] = 0.0f; }

  int kendW = srow0 + 15;
  int kendB = srowB + 127;

  stage(0, 0);
  wait_async0();
  __syncthreads();
  int p = 0;
  for (int kb = 0; kb <= kendB; kb += 32) {
    if (kb + 32 <= kendB) stage(kb + 32, p ^ 1);
    if (kb <= kendW) {
      v8f sc0 = {}, sc1 = {};
#pragma unroll
      for (int c = 0; c < 4; ++c) {
        v16h b0 = frag16(&ks[p][n][c * 32 + half * 16]);
        v16h b1 = frag16(&ks[p][16 + n][c * 32 + half * 16]);
        sc0 = wmma_f16(qa[c], b0, sc0);
        sc1 = wmma_f16(qa[c], b1, sc1);
      }
      float sarr[2][8];
#pragma unroll
      for (int r = 0; r < 8; ++r) {
        int rowg = srow0 + r + 8 * half;
        float s0 = sc0[r] * SCALEc; if (kb + n      > rowg) s0 = -1.0e30f;
        float s1 = sc1[r] * SCALEc; if (kb + 16 + n > rowg) s1 = -1.0e30f;
        sarr[0][r] = s0; sarr[1][r] = s1;
      }
      float rm[8];
#pragma unroll
      for (int r = 0; r < 8; ++r) rm[r] = fmaxf(sarr[0][r], sarr[1][r]);
#pragma unroll
      for (int xm = 1; xm < 16; xm <<= 1)
#pragma unroll
        for (int r = 0; r < 8; ++r) rm[r] = fmaxf(rm[r], __shfl_xor(rm[r], xm, 32));
      float alpha[8];
#pragma unroll
      for (int r = 0; r < 8; ++r) {
        float mn = fmaxf(m8[r], rm[r]);
        alpha[r] = __expf(m8[r] - mn);
        m8[r] = mn;
      }
      float rs[8];
#pragma unroll
      for (int r = 0; r < 8; ++r) {
        float p0 = __expf(sarr[0][r] - m8[r]);
        float p1 = __expf(sarr[1][r] - m8[r]);
        rs[r] = p0 + p1;
        pbuf[w][r + 8 * half][n]      = (_Float16)p0;
        pbuf[w][r + 8 * half][16 + n] = (_Float16)p1;
      }
#pragma unroll
      for (int xm = 1; xm < 16; xm <<= 1)
#pragma unroll
        for (int r = 0; r < 8; ++r) rs[r] += __shfl_xor(rs[r], xm, 32);
#pragma unroll
      for (int r = 0; r < 8; ++r) l8[r] = l8[r] * alpha[r] + rs[r];
#pragma unroll
      for (int dc = 0; dc < 8; ++dc)
#pragma unroll
        for (int r = 0; r < 8; ++r) acc[dc][r] *= alpha[r];
      v16h pa = loadA16(&pbuf[w][0][0], 40, lane);
#pragma unroll
      for (int dc = 0; dc < 8; ++dc) {
        v16h vb = frag16(&vs[p][dc * 16 + n][half * 16]);
        acc[dc] = wmma_f16(pa, vb, acc[dc]);
      }
    }
    wait_async0();          // async K copies done before the buffer handoff
    __syncthreads();
    p ^= 1;
  }

#pragma unroll
  for (int r = 0; r < 8; ++r) {
    float invl = 1.0f / l8[r];
    int s = srow0 + r + 8 * half;
    size_t rowoff = ((size_t)b * Sc + s) * (Hc * Dc) + (size_t)h * Dc;
#pragma unroll
    for (int dc = 0; dc < 8; ++dc)
      ao[rowoff + dc * 16 + n] = (_Float16)(acc[dc][r] * invl);
  }
}

// ==================== Kernel 4: output projection ====================
__global__ __launch_bounds__(256) void out_gemm_k(
    const _Float16* __restrict__ ao, const float* __restrict__ wo,
    float* __restrict__ out) {
  __shared__ __align__(16) _Float16 bs[2][128][40];
  int lane = threadIdx.x, w = threadIdx.y;
  int tid  = w * 32 + lane;
  int mt   = blockIdx.x >> 4;       // 32 row-tiles of 128
  int nb   = blockIdx.x & 15;       // 16 col-tiles of 128
  int rowW = mt * 128 + w * 16;
  int colb = nb * 128;

  int skk = (tid >> 5) * 4;
  int sn4 = (tid & 31) * 4;
  auto stage = [&](int k0, int p) {
#pragma unroll
    for (int rp = 0; rp < 2; ++rp) {
      int kr = skk + rp * 2;
      v4f r0 = *(const v4f*)(wo + (size_t)(k0 + kr) * 2048 + colb + sn4);
      v4f r1 = *(const v4f*)(wo + (size_t)(k0 + kr + 1) * 2048 + colb + sn4);
#pragma unroll
      for (int i = 0; i < 4; ++i) {
        v2h pk = { (_Float16)r0[i], (_Float16)r1[i] };
        *(v2h*)&bs[p][sn4 + i][kr] = pk;
      }
    }
  };

  v8f acc[8] = {};
  stage(0, 0);
  __syncthreads();
  int p = 0;
  int n = lane & 15, half = lane >> 4;
  for (int k0 = 0; k0 < 2048; k0 += 32) {
    if (k0 + 32 < 2048) stage(k0 + 32, p ^ 1);
    v16h a = loadA16(ao + (size_t)rowW * 2048 + k0, 2048, lane);
    v16h bf[8];
#pragma unroll
    for (int t = 0; t < 8; ++t) bf[t] = frag16(&bs[p][t * 16 + n][half * 16]);
#pragma unroll
    for (int t = 0; t < 8; ++t) acc[t] = wmma_f16(a, bf[t], acc[t]);
    __syncthreads();
    p ^= 1;
  }

#pragma unroll
  for (int t = 0; t < 8; ++t) {
    int col = colb + t * 16 + n;
#pragma unroll
    for (int r = 0; r < 8; ++r) {
      int row = rowW + r + 8 * half;
      out[(size_t)row * 2048 + col] = acc[t][r];
    }
  }
}

// ==================== host launch ====================
extern "C" void kernel_launch(void* const* d_in, const int* in_sizes, int n_in,
                              void* d_out, int out_size, void* d_ws, size_t ws_size,
                              hipStream_t stream) {
  const float* x    = (const float*)d_in[0];
  const float* wq   = (const float*)d_in[1];
  const float* wk   = (const float*)d_in[2];
  const float* wv   = (const float*)d_in[3];
  const float* wo   = (const float*)d_in[4];
  const float* qg   = (const float*)d_in[5];
  const float* kg   = (const float*)d_in[6];
  const float* cosc = (const float*)d_in[7];
  const float* sinc = (const float*)d_in[8];
  float* out = (float*)d_out;

  char* ws = (char*)d_ws;
  float*    qf32 = (float*)(ws + QF32_OFF);
  float*    kf32 = (float*)(ws + KF32_OFF);
  _Float16* qf16 = (_Float16*)(ws + QF16_OFF);
  _Float16* kf16 = (_Float16*)(ws + KF16_OFF);
  _Float16* vf16 = (_Float16*)(ws + VF16_OFF);
  _Float16* aof  = (_Float16*)(ws + AO_OFF);

  dim3 blk(32, 8);
  qkv_gemm_k<<<32 * 32, blk, 0, stream>>>(x, wq, wk, wv, qf32, kf32, vf16);
  norm_rope_k<<<12288, blk, 0, stream>>>(qf32, kf32, qg, kg, cosc, sinc, qf16, kf16);
  attn_k<<<dim3(16, Hc, Bc), blk, 0, stream>>>(qf16, kf16, vf16, aof);
  out_gemm_k<<<32 * 16, blk, 0, stream>>>(aof, wo, out);

  (void)in_sizes; (void)n_in; (void)out_size; (void)ws_size;
}
